// PerceiverResampler_18829136626078
// MI455X (gfx1250) — matrix operand order repaired and checked by reading, not compiled, not run on hardware
//
#include <hip/hip_runtime.h>
#include <hip/hip_bf16.h>
#include <math.h>

// ---------------------------------------------------------------------------
// PerceiverResampler for MI455X (gfx1250, wave32).
// Large GEMMs: v_wmma_f32_16x16x32_f16. The B slab (shared by all 8 waves of
// the block) is staged into LDS with global_load_async_to_lds_b128 (ASYNCcnt),
// double-buffered in 64-K slabs (8KB each) -> one barrier per 8 WMMAs.
// A is streamed per-wave with global_load_b128 + prefetch.
// Attention / LN / softmax are f32 VALU (tiny FLOP share).
// ---------------------------------------------------------------------------

typedef __attribute__((ext_vector_type(16))) _Float16 v16h;
typedef __attribute__((ext_vector_type(8)))  float    v8f;

union H16 { uint4 u[2]; v16h v; };

// One lane moves 16B global -> LDS through the async unit (no VGPR data path).
__device__ __forceinline__ void async_ld_b128(unsigned ldsOff, const void* gsrc)
{
  unsigned long long ga = (unsigned long long)gsrc;
  asm volatile("global_load_async_to_lds_b128 %0, %1, off"
               :: "v"(ldsOff), "v"(ga) : "memory");
}
__device__ __forceinline__ void wait_async0()
{
  asm volatile("s_wait_asynccnt 0x0" ::: "memory");
}

// --------------------------- WMMA GEMM -------------------------------------
// C[M,N](f32) = A[M,K](f16 row major) * Bt[N,K](f16 row major == B^T)
// Output row remap: orow = (m / grpIn) * grpOut + (m % grpIn) + rowOff.
// accumulate: C += result.  Requires M%128==0, N%64==0, K%64==0 (all callers).
//
// Lane layouts used by this kernel (wave32, contiguous K order):
//  A 16x32 f16 : lanes 0-15 row m=lane, K=k..k+15;
//                lanes 16-31 row m=lane-16, K=k+16..k+31.
//  B 32x16 f16 : lanes 0-15 col n=lane, K=k..k+15; lanes 16-31 K=k+16..k+31.
//  C 16x16 f32 : vgpr r: lanes 0-15 -> (M=r, N=lane); lanes 16-31 -> (M=r+8).
__device__ __forceinline__ void store_tile(const v8f& acc, float* __restrict__ C,
                                           int rowBase, int mHi, int col, int M,
                                           int grpIn, int grpOut, int rowOff,
                                           int ldc, int accumulate)
{
#pragma unroll
  for (int r = 0; r < 8; ++r) {
    int m = rowBase + mHi + r;
    if (m < M) {
      int orow = (m / grpIn) * grpOut + (m % grpIn) + rowOff;
      size_t idx = (size_t)orow * (size_t)ldc + (size_t)col;
      float v = acc[r];
      if (accumulate) v += C[idx];
      C[idx] = v;
    }
  }
}

__global__ __launch_bounds__(256) void k_gemm_f16(
    const _Float16* __restrict__ A, const _Float16* __restrict__ Bt,
    float* __restrict__ C, int M, int N, int K,
    int grpIn, int grpOut, int rowOff, int ldc, int accumulate)
{
  // Double-buffered B slab: 64 cols x 64 K halves = 8KB per buffer.
  // Layout: col-major slabs, col c at byte c*128, K-halves packed in order.
  __shared__ _Float16 bs[2][64 * 64];

  const int tid  = threadIdx.x;
  const int wave = tid >> 5;                 // 8 waves / block
  const int lane = tid & 31;
  const int rowBase = blockIdx.x * 128 + wave * 16;
  const int colBase = blockIdx.y * 64;
  const int l16 = lane & 15;
  const int hi  = lane >> 4;

  // A stream (per-wave rows, unique -> direct global b128).
  const _Float16* aPtr = A + (size_t)(rowBase + l16) * (size_t)K + (hi ? 16 : 0);

  // B staging: thread t moves 32 contiguous bytes of col (t>>2), K chunk (t&3)*16.
  const _Float16* stageSrc = Bt + (size_t)(colBase + (tid >> 2)) * (size_t)K
                                + (size_t)((tid & 3) * 16);
  unsigned dst0 = (unsigned)(size_t)(void*)&bs[0][0] + (unsigned)(tid * 32);
  unsigned dst1 = (unsigned)(size_t)(void*)&bs[1][0] + (unsigned)(tid * 32);

  v8f acc0 = {}, acc1 = {}, acc2 = {}, acc3 = {};

  // prologue: slab 0 -> buf 0
  async_ld_b128(dst0, stageSrc);
  async_ld_b128(dst0 + 16, stageSrc + 8);

  int buf = 0;
  for (int kk = 0; kk < K; kk += 64, buf ^= 1) {
    wait_async0();                           // own async writes of current buf done
    __syncthreads();                         // slab visible; prev readers of other buf done
    if (kk + 64 < K) {
      unsigned nd = buf ? dst0 : dst1;
      const _Float16* ns = stageSrc + kk + 64;
      async_ld_b128(nd, ns);
      async_ld_b128(nd + 16, ns + 8);
      __builtin_prefetch(aPtr + kk + 64, 0, 3);   // next A slab toward near cache
    }

    const uint4* bsv = (const uint4*)&bs[buf][0]; // 16B units; col = 8 units
#pragma unroll
    for (int s = 0; s < 2; ++s) {            // two 32-K WMMA steps per slab
      H16 a;
      a.u[0] = *(const uint4*)(aPtr + kk + s * 32);
      a.u[1] = *(const uint4*)(aPtr + kk + s * 32 + 8);

      const int bi = l16 * 8 + s * 4 + hi * 2;
      H16 b0, b1, b2, b3;
      b0.u[0] = bsv[bi];       b0.u[1] = bsv[bi + 1];
      b1.u[0] = bsv[bi + 128]; b1.u[1] = bsv[bi + 129];
      b2.u[0] = bsv[bi + 256]; b2.u[1] = bsv[bi + 257];
      b3.u[0] = bsv[bi + 384]; b3.u[1] = bsv[bi + 385];

      acc0 = __builtin_amdgcn_wmma_f32_16x16x32_f16(false, a.v, false, b0.v, (short)0, acc0, false, false);
      acc1 = __builtin_amdgcn_wmma_f32_16x16x32_f16(false, a.v, false, b1.v, (short)0, acc1, false, false);
      acc2 = __builtin_amdgcn_wmma_f32_16x16x32_f16(false, a.v, false, b2.v, (short)0, acc2, false, false);
      acc3 = __builtin_amdgcn_wmma_f32_16x16x32_f16(false, a.v, false, b3.v, (short)0, acc3, false, false);
    }
  }

  const int mHi = hi * 8;
  store_tile(acc0, C, rowBase, mHi, colBase + 0  + l16, M, grpIn, grpOut, rowOff, ldc, accumulate);
  store_tile(acc1, C, rowBase, mHi, colBase + 16 + l16, M, grpIn, grpOut, rowOff, ldc, accumulate);
  store_tile(acc2, C, rowBase, mHi, colBase + 32 + l16, M, grpIn, grpOut, rowOff, ldc, accumulate);
  store_tile(acc3, C, rowBase, mHi, colBase + 48 + l16, M, grpIn, grpOut, rowOff, ldc, accumulate);
}

// ------------------- weight convert + transpose (f32 KxN -> f16 NxK) -------
__global__ __launch_bounds__(256) void k_cvtT(const float* __restrict__ W,
                                              _Float16* __restrict__ Wt,
                                              int K, int N)
{
  size_t idx = (size_t)blockIdx.x * 256 + threadIdx.x;
  if (idx >= (size_t)K * (size_t)N) return;
  int n = (int)(idx % (size_t)N);
  int k = (int)(idx / (size_t)N);
  Wt[(size_t)n * (size_t)K + (size_t)k] = (_Float16)W[idx];
}

// ------------------------------- LayerNorm ---------------------------------
__global__ __launch_bounds__(256) void k_ln(const float* __restrict__ X,
                                            const float* __restrict__ g,
                                            const float* __restrict__ b,
                                            _Float16* __restrict__ outH,
                                            float* __restrict__ outF)
{
  const int D = 2048;
  size_t row = blockIdx.x;
  const float* x = X + row * (size_t)D;
  __shared__ float red[256];
  int tid = threadIdx.x;
  float s = 0.f;
  for (int i = tid; i < D; i += 256) s += x[i];
  red[tid] = s; __syncthreads();
  for (int o = 128; o > 0; o >>= 1) { if (tid < o) red[tid] += red[tid + o]; __syncthreads(); }
  float mean = red[0] * (1.0f / D);
  __syncthreads();
  float q = 0.f;
  for (int i = tid; i < D; i += 256) { float d = x[i] - mean; q += d * d; }
  red[tid] = q; __syncthreads();
  for (int o = 128; o > 0; o >>= 1) { if (tid < o) red[tid] += red[tid + o]; __syncthreads(); }
  float inv = rsqrtf(red[0] * (1.0f / D) + 1e-5f);
  for (int i = tid; i < D; i += 256) {
    float y = (x[i] - mean) * inv * g[i] + b[i];
    if (outH) outH[row * (size_t)D + i] = (_Float16)y;
    if (outF) outF[row * (size_t)D + i] = y;
  }
}

// -------------------- grouped attention (49 keys, 8 heads) -----------------
__global__ __launch_bounds__(64) void k_attn_group(const float* __restrict__ Q,
                                                   const float* __restrict__ KV,
                                                   _Float16* __restrict__ O)
{
  const int NG = 49, DH = 64;
  int idx = blockIdx.x;
  int h = idx & 7;
  int t = (idx >> 3) % 24;
  int b = idx / (8 * 24);
  size_t row0 = (size_t)b * 1176 + (size_t)t * NG;
  const float* qb = Q + row0 * 512 + h * DH;
  const float* kb = KV + row0 * 1024 + h * DH;
  const float* vb = KV + row0 * 1024 + 512 + h * DH;
  _Float16* ob = O + row0 * 512 + h * DH;

  __shared__ float ks[49 * 64];
  __shared__ float vs[49 * 64];
  __shared__ float p[64];
  __shared__ float red[64];

  int tid = threadIdx.x;
  for (int e = tid; e < NG * DH; e += 64) {
    int j = e >> 6, d = e & 63;
    ks[e] = kb[(size_t)j * 1024 + d];
    vs[e] = vb[(size_t)j * 1024 + d];
  }
  __syncthreads();

  for (int i = 0; i < NG; ++i) {
    const float* qi = qb + (size_t)i * 512;
    float sim = -3.0e38f;
    if (tid < NG) {
      float s = 0.f;
      for (int d = 0; d < DH; ++d) s += qi[d] * ks[tid * DH + d];
      sim = s * 0.125f;
    }
    red[tid] = sim; __syncthreads();
    for (int o = 32; o > 0; o >>= 1) { if (tid < o) red[tid] = fmaxf(red[tid], red[tid + o]); __syncthreads(); }
    float mx = red[0]; __syncthreads();
    float e = (tid < NG) ? __expf(sim - mx) : 0.f;
    p[tid] = e;
    red[tid] = e; __syncthreads();
    for (int o = 32; o > 0; o >>= 1) { if (tid < o) red[tid] += red[tid + o]; __syncthreads(); }
    float inv = 1.0f / red[0]; __syncthreads();
    float acc = 0.f;
    for (int j = 0; j < NG; ++j) acc += p[j] * vs[j * DH + tid];
    ob[(size_t)i * 512 + tid] = (_Float16)(acc * inv);
    __syncthreads();
  }
}

// ----------------- perceiver cross-attention (1432 keys) -------------------
__global__ __launch_bounds__(256) void k_sim(const float* __restrict__ Q,
                                             const float* __restrict__ KV,
                                             float* __restrict__ S)
{
  const int LK = 1432;
  int bh = blockIdx.z;
  int b = bh >> 3, h = bh & 7;
  int i = blockIdx.y;
  __shared__ float qs[64];
  if (threadIdx.x < 64)
    qs[threadIdx.x] = Q[((size_t)b * 256 + i) * 512 + h * 64 + threadIdx.x] * 0.125f;
  __syncthreads();
  int j = blockIdx.x * 256 + threadIdx.x;
  if (j >= LK) return;
  const float* kr = KV + ((size_t)b * LK + j) * 1024 + h * 64;
  float s = 0.f;
#pragma unroll
  for (int d = 0; d < 64; ++d) s += qs[d] * kr[d];
  S[((size_t)bh * 256 + i) * LK + j] = s;
}

__global__ __launch_bounds__(256) void k_softmax(float* __restrict__ S)
{
  const int LK = 1432;
  float* s = S + (size_t)blockIdx.x * LK;
  __shared__ float red[256];
  int tid = threadIdx.x;
  float mx = -3.0e38f;
  for (int j = tid; j < LK; j += 256) mx = fmaxf(mx, s[j]);
  red[tid] = mx; __syncthreads();
  for (int o = 128; o > 0; o >>= 1) { if (tid < o) red[tid] = fmaxf(red[tid], red[tid + o]); __syncthreads(); }
  mx = red[0]; __syncthreads();
  float sum = 0.f;
  for (int j = tid; j < LK; j += 256) { float e = __expf(s[j] - mx); s[j] = e; sum += e; }
  red[tid] = sum; __syncthreads();
  for (int o = 128; o > 0; o >>= 1) { if (tid < o) red[tid] += red[tid + o]; __syncthreads(); }
  float inv = 1.0f / red[0]; __syncthreads();
  for (int j = tid; j < LK; j += 256) s[j] *= inv;
}

__global__ __launch_bounds__(64) void k_av(const float* __restrict__ S,
                                           const float* __restrict__ KV,
                                           _Float16* __restrict__ O)
{
  const int LK = 1432;
  int bh = blockIdx.y;
  int b = bh >> 3, h = bh & 7;
  int i = blockIdx.x;
  const float* p = S + ((size_t)bh * 256 + i) * LK;
  const float* vb = KV + (size_t)b * LK * 1024 + 512 + h * 64 + threadIdx.x;
  float acc = 0.f;
  for (int j = 0; j < LK; ++j) acc += p[j] * vb[(size_t)j * 1024];
  O[((size_t)b * 256 + i) * 512 + h * 64 + threadIdx.x] = (_Float16)acc;
}

// ------------------------------ elementwise --------------------------------
__global__ __launch_bounds__(256) void k_addseg(float* __restrict__ X,
                                                const float* __restrict__ SEG)
{
  size_t idx = (size_t)blockIdx.x * 256 + threadIdx.x;
  if (idx >= (size_t)18816 * 2048) return;
  int d = (int)(idx & 2047);
  size_t r = idx >> 11;
  int t = (int)((r % 1176) / 49);
  X[idx] += SEG[(size_t)t * 2048 + d];
}

__global__ __launch_bounds__(256) void k_bcast_lat(const float* __restrict__ L,
                                                   float* __restrict__ LAT)
{
  size_t idx = (size_t)blockIdx.x * 256 + threadIdx.x;
  if (idx >= (size_t)16 * 256 * 2048) return;
  LAT[idx] = L[idx % ((size_t)256 * 2048)];
}

__global__ __launch_bounds__(256) void k_gelu(const float* __restrict__ H,
                                              _Float16* __restrict__ G, size_t n)
{
  size_t idx = (size_t)blockIdx.x * 256 + threadIdx.x;
  if (idx >= n) return;
  float x = H[idx];
  float y = 0.5f * x * (1.0f + erff(x * 0.70710678118654752f));
  G[idx] = (_Float16)y;
}

// ------------------------------- driver ------------------------------------
extern "C" void kernel_launch(void* const* d_in, const int* in_sizes, int n_in,
                              void* d_out, int out_size, void* d_ws, size_t ws_size,
                              hipStream_t stream)
{
  (void)in_sizes; (void)n_in; (void)out_size; (void)ws_size;
  const float* x       = (const float*)d_in[0];
  const float* gns     = (const float*)d_in[1];
  const float* gnb     = (const float*)d_in[2];
  const float* g_wq    = (const float*)d_in[3];
  const float* g_wkv   = (const float*)d_in[4];
  const float* g_wo    = (const float*)d_in[5];
  const float* seg     = (const float*)d_in[6];
  const float* latents = (const float*)d_in[7];
  const float* a_nm_s  = (const float*)d_in[8];
  const float* a_nm_b  = (const float*)d_in[9];
  const float* a_nl_s  = (const float*)d_in[10];
  const float* a_nl_b  = (const float*)d_in[11];
  const float* a_wq    = (const float*)d_in[12];
  const float* a_wkv   = (const float*)d_in[13];
  const float* a_wo    = (const float*)d_in[14];
  const float* f_n_s   = (const float*)d_in[15];
  const float* f_n_b   = (const float*)d_in[16];
  const float* f_w1    = (const float*)d_in[17];
  const float* f_w2    = (const float*)d_in[18];
  const float* o_n_s   = (const float*)d_in[19];
  const float* o_n_b   = (const float*)d_in[20];
  float* out = (float*)d_out;

  const int MTOT = 18816, DIM = 2048, INNER = 512, NLAT = 256, DFF = 8192;
  const int MLAT = 16 * NLAT;                  // 4096

  char* wsB = (char*)d_ws;
  size_t off = 0;
  auto alloc = [&](size_t bytes) -> char* {
    char* p = wsB + off;
    off += (bytes + 255) & ~(size_t)255;
    return p;
  };

  _Float16* WT_gwq  = (_Float16*)alloc((size_t)INNER * DIM * 2);
  _Float16* WT_gwkv = (_Float16*)alloc((size_t)2 * INNER * DIM * 2);
  _Float16* WT_gwo  = (_Float16*)alloc((size_t)DIM * INNER * 2);
  _Float16 *WT_awq[3], *WT_awkv[3], *WT_awo[3], *WT_fw1[3], *WT_fw2[3];
  for (int l = 0; l < 3; ++l) {
    WT_awq[l]  = (_Float16*)alloc((size_t)INNER * DIM * 2);
    WT_awkv[l] = (_Float16*)alloc((size_t)2 * INNER * DIM * 2);
    WT_awo[l]  = (_Float16*)alloc((size_t)DIM * INNER * 2);
    WT_fw1[l]  = (_Float16*)alloc((size_t)DFF * DIM * 2);
    WT_fw2[l]  = (_Float16*)alloc((size_t)DIM * DFF * 2);
  }
  _Float16* xh   = (_Float16*)alloc((size_t)MTOT * DIM * 2);
  float*    xbuf = (float*)   alloc((size_t)MTOT * DIM * 4);
  float*    lat  = (float*)   alloc((size_t)MLAT * DIM * 4);
  _Float16* lmh  = (_Float16*)alloc((size_t)MLAT * DIM * 2);
  _Float16* hnh  = (_Float16*)alloc((size_t)MLAT * DIM * 2);

  char* rA = alloc(134873088);
  float*    qg  = (float*)rA;                        // 18816x512 f32
  float*    kvg = (float*)(rA + 38535168);           // 18816x1024 f32
  _Float16* ogh = (_Float16*)(rA + 115605504);       // 18816x512 f16
  float*    q2  = (float*)rA;                        // 4096x512 f32
  float*    kv2 = (float*)(rA + 8388608);            // 16x1432x1024 f32
  _Float16* o2h = (_Float16*)(rA + 102236160);       // 4096x512 f16
  char* rB = alloc(201326592);
  float*    sim  = (float*)rB;                       // 128x256x1432 f32
  float*    hbuf = (float*)rB;                       // 4096x8192 f32
  _Float16* gh   = (_Float16*)(rB + 134217728);      // 4096x8192 f16

  auto cvt = [&](const float* W, _Float16* Wt, int K, int N) {
    size_t tot = (size_t)K * N;
    k_cvtT<<<dim3((unsigned)((tot + 255) / 256)), dim3(256), 0, stream>>>(W, Wt, K, N);
  };
  auto gemm = [&](const _Float16* A, const _Float16* Bt, float* C, int M, int N, int K,
                  int gi, int go, int ro, int ldc, int acc) {
    dim3 grid((unsigned)(M / 128), (unsigned)(N / 64));   // all M % 128 == 0
    k_gemm_f16<<<grid, 256, 0, stream>>>(A, Bt, C, M, N, K, gi, go, ro, ldc, acc);
  };

  // ---- weight prep ----
  cvt(g_wq, WT_gwq, DIM, INNER);
  cvt(g_wkv, WT_gwkv, DIM, 2 * INNER);
  cvt(g_wo, WT_gwo, INNER, DIM);
  for (int l = 0; l < 3; ++l) {
    cvt(a_wq  + (size_t)l * DIM * INNER,     WT_awq[l],  DIM, INNER);
    cvt(a_wkv + (size_t)l * DIM * 2 * INNER, WT_awkv[l], DIM, 2 * INNER);
    cvt(a_wo  + (size_t)l * INNER * DIM,     WT_awo[l],  INNER, DIM);
    cvt(f_w1  + (size_t)l * DIM * DFF,       WT_fw1[l],  DIM, DFF);
    cvt(f_w2  + (size_t)l * DFF * DIM,       WT_fw2[l],  DFF, DIM);
  }

  // ---- stage 1: grouped self-attention over x ----
  k_ln<<<MTOT, 256, 0, stream>>>(x, gns, gnb, xh, (float*)nullptr);
  gemm(xh, WT_gwq,  qg,  MTOT, INNER,     DIM, 1, 1, 0, INNER,     0);
  gemm(xh, WT_gwkv, kvg, MTOT, 2 * INNER, DIM, 1, 1, 0, 2 * INNER, 0);
  k_attn_group<<<16 * 24 * 8, 64, 0, stream>>>(qg, kvg, ogh);
  gemm(ogh, WT_gwo, xbuf, MTOT, DIM, INNER, 1, 1, 0, DIM, 0);
  k_addseg<<<(unsigned)(((size_t)MTOT * DIM + 255) / 256), 256, 0, stream>>>(xbuf, seg);
  k_bcast_lat<<<(unsigned)(((size_t)MLAT * DIM + 255) / 256), 256, 0, stream>>>(latents, lat);

  // ---- perceiver layers ----
  for (int l = 0; l < 3; ++l) {
    k_ln<<<MTOT, 256, 0, stream>>>(xbuf, a_nm_s + l * DIM, a_nm_b + l * DIM, xh, (float*)nullptr);
    k_ln<<<MLAT, 256, 0, stream>>>(lat,  a_nl_s + l * DIM, a_nl_b + l * DIM, lmh, (float*)nullptr);
    gemm(lmh, WT_awq[l], q2, MLAT, INNER, DIM, 1, 1, 0, INNER, 0);
    gemm(xh,  WT_awkv[l], kv2, MTOT, 2 * INNER, DIM, 1176, 1432, 0,    2 * INNER, 0);
    gemm(lmh, WT_awkv[l], kv2, MLAT, 2 * INNER, DIM, 256,  1432, 1176, 2 * INNER, 0);
    k_sim<<<dim3(6, NLAT, 128), 256, 0, stream>>>(q2, kv2, sim);
    k_softmax<<<128 * NLAT, 256, 0, stream>>>(sim);
    k_av<<<dim3(NLAT, 128), 64, 0, stream>>>(sim, kv2, o2h);
    gemm(o2h, WT_awo[l], lat, MLAT, DIM, INNER, 1, 1, 0, DIM, 1);
    k_ln<<<MLAT, 256, 0, stream>>>(lat, f_n_s + l * DIM, f_n_b + l * DIM, hnh, (float*)nullptr);
    gemm(hnh, WT_fw1[l], hbuf, MLAT, DFF, DIM, 1, 1, 0, DFF, 0);
    k_gelu<<<(unsigned)(((size_t)MLAT * DFF + 255) / 256), 256, 0, stream>>>(hbuf, gh, (size_t)MLAT * DFF);
    gemm(gh, WT_fw2[l], lat, MLAT, DIM, DFF, 1, 1, 0, DIM, 1);
  }

  // ---- output LN -> f32 ----
  k_ln<<<MLAT, 256, 0, stream>>>(lat, o_n_s, o_n_b, (_Float16*)nullptr, out);
}
